// EdgeUpdate_18614388260892
// MI455X (gfx1250) — matrix operand order repaired and checked by reading, not compiled, not run on hardware
//
#include <hip/hip_runtime.h>
#include <hip/hip_bf16.h>

typedef __attribute__((ext_vector_type(16))) _Float16 v16h;
typedef __attribute__((ext_vector_type(8)))  _Float16 v8h;
typedef __attribute__((ext_vector_type(4)))  _Float16 v4h;
typedef __attribute__((ext_vector_type(8)))  float    v8f;

#define D_NODE 128
#define D_EDGE 128
#define RBF    16
#define IN_DIM 400
#define K1PAD  416          // 13 chunks of 32
#define K1C    13
#define K2C    4
#define M_TILE 64
#define XSTR   424          // halves; 848B row -> 212 dwords == 20 mod 64 -> conflict-free
#define HSTR   136          // halves; 272B row
#define YSTR   132          // floats; 528B row
#define W1P_HALVES (K1C*8*512)   // 53248 halves = 106496 B
#define W2P_HALVES (K2C*8*512)   // 16384 halves =  32768 B

// ---------------- weight pre-pack: f32 row-major -> f16 fragment-native -----
// B fragment (32x16 f16): lane L holds column n = L&15; half = L>>4;
// j-th half (j=0..15) is k = kc*32 + half*8 + (j&7) + (j>>3)*16. Lane data contiguous (32B).
__global__ void pack_w1_kernel(const float* __restrict__ W1, _Float16* __restrict__ out) {
    int idx = blockIdx.x * 256 + threadIdx.x;
    if (idx >= W1P_HALVES) return;
    int j   = idx & 15;
    int L   = (idx >> 4) & 31;
    int blk = idx >> 9;          // kc*8 + nt
    int kc  = blk >> 3;
    int nt  = blk & 7;
    int n   = nt * 16 + (L & 15);
    int k   = kc * 32 + (L >> 4) * 8 + (j & 7) + (j >> 3) * 16;
    float v = (k < IN_DIM) ? W1[k * D_EDGE + n] : 0.0f;
    out[idx] = (_Float16)v;
}

__global__ void pack_w2_kernel(const float* __restrict__ W2, _Float16* __restrict__ out) {
    int idx = blockIdx.x * 256 + threadIdx.x;
    if (idx >= W2P_HALVES) return;
    int j   = idx & 15;
    int L   = (idx >> 4) & 31;
    int blk = idx >> 9;
    int kc  = blk >> 3;
    int nt  = blk & 7;
    int n   = nt * 16 + (L & 15);
    int k   = kc * 32 + (L >> 4) * 8 + (j & 7) + (j >> 3) * 16;
    out[idx] = (_Float16)W2[k * D_EDGE + n];
}

__device__ __forceinline__ float silu_f(float x) {
    return x * __builtin_amdgcn_rcpf(1.0f + __expf(-x));
}

// ---------------- main fused kernel ----------------------------------------
__global__ __launch_bounds__(256, 2)
void edge_mlp_kernel(const float* __restrict__ ns, const float* __restrict__ ef,
                     const float* __restrict__ dptr,
                     const int* __restrict__ srci, const int* __restrict__ dsti,
                     const _Float16* __restrict__ w1p, const _Float16* __restrict__ w2p,
                     const float* __restrict__ b1, const float* __restrict__ b2,
                     const float* __restrict__ gamma, const float* __restrict__ beta,
                     float* __restrict__ out) {
    __shared__ __align__(16) unsigned char smem[M_TILE * XSTR * 2]; // 54272 B
    __shared__ int sidx[M_TILE];
    __shared__ int didx[M_TILE];

    _Float16* Xl = (_Float16*)smem;                       // [64][424] f16, phase 1
    _Float16* Hl = (_Float16*)smem;                       // [64][136] f16, aliases X
    float*    Yl = (float*)(smem + M_TILE * HSTR * 2);    // [64][132] f32, after H

    const int tid    = threadIdx.x;
    const int L      = tid & 31;
    const int wv     = tid >> 5;
    const int strip  = wv & 3;        // row strip: rows strip*16 .. +16
    const int nhalf  = wv >> 2;       // N half: cols nhalf*64 .. +64
    const int lane15 = L & 15;
    const int halfL  = L >> 4;
    const int e0     = blockIdx.x * M_TILE;

    // ---- phase 0: stage gather indices ----
    if (tid < M_TILE) {
        sidx[tid] = srci[e0 + tid];
        didx[tid] = dsti[e0 + tid];
    }
    __syncthreads();

    // ---- phase 1: build X tile (concat + f32->f16), 104 quads per row ----
    for (int i = 0; i < 26; ++i) {
        int qid = tid + i * 256;          // < 64*104
        int r   = qid / 104;
        int c   = (qid - r * 104) * 4;    // padded column
        float4 v;
        if (c < 128) {
            v = *(const float4*)(ns + (size_t)sidx[r] * D_NODE + c);
        } else if (c < 256) {
            v = *(const float4*)(ns + (size_t)didx[r] * D_NODE + (c - 128));
        } else if (c < 384) {
            v = *(const float4*)(ef + (size_t)(e0 + r) * D_EDGE + (c - 256));
        } else if (c < 400) {
            v = *(const float4*)(dptr + (size_t)(e0 + r) * RBF + (c - 384));
        } else {
            v = make_float4(0.f, 0.f, 0.f, 0.f);
        }
        v4h h4 = { (_Float16)v.x, (_Float16)v.y, (_Float16)v.z, (_Float16)v.w };
        *(v4h*)(Xl + r * XSTR + c) = h4;
    }
    __syncthreads();

    // ---- phase 2: layer 1 GEMM, K = 416 (13 chunks) ----
    v8f acc[4];
    float b1v[4];
#pragma unroll
    for (int nt = 0; nt < 4; ++nt) {
        acc[nt] = (v8f){};
        b1v[nt] = b1[nhalf * 64 + nt * 16 + lane15];
    }
    const _Float16* Xrow = Xl + (strip * 16 + lane15) * XSTR;
    for (int kc = 0; kc < K1C; ++kc) {
        v8h lo = *(const v8h*)(Xrow + kc * 32 + halfL * 8);
        v8h hi = *(const v8h*)(Xrow + kc * 32 + 16 + halfL * 8);
        v16h a = __builtin_shufflevector(lo, hi, 0,1,2,3,4,5,6,7,8,9,10,11,12,13,14,15);
#pragma unroll
        for (int nt = 0; nt < 4; ++nt) {
            const v16h b = *(const v16h*)(w1p + (size_t)(kc * 8 + nhalf * 4 + nt) * 512 + L * 16);
            acc[nt] = __builtin_amdgcn_wmma_f32_16x16x32_f16(
                false, a, false, b, (short)0, acc[nt], false, false);
        }
    }
    __syncthreads();   // everyone done reading X before H aliases it

    // ---- bias + SiLU, store H (f16) ----
#pragma unroll
    for (int nt = 0; nt < 4; ++nt) {
        int ncol = nhalf * 64 + nt * 16 + lane15;
#pragma unroll
        for (int v = 0; v < 8; ++v) {
            float s = silu_f(acc[nt][v] + b1v[nt]);
            Hl[(strip * 16 + v + 8 * halfL) * HSTR + ncol] = (_Float16)s;
        }
    }
    __syncthreads();

    // ---- phase 3: layer 2 GEMM, K = 128 (4 chunks) ----
    v8f acc2[4];
    float b2v[4];
#pragma unroll
    for (int nt = 0; nt < 4; ++nt) {
        acc2[nt] = (v8f){};
        b2v[nt] = b2[nhalf * 64 + nt * 16 + lane15];
    }
    const _Float16* Hrow = Hl + (strip * 16 + lane15) * HSTR;
#pragma unroll
    for (int kc = 0; kc < K2C; ++kc) {
        v8h lo = *(const v8h*)(Hrow + kc * 32 + halfL * 8);
        v8h hi = *(const v8h*)(Hrow + kc * 32 + 16 + halfL * 8);
        v16h a = __builtin_shufflevector(lo, hi, 0,1,2,3,4,5,6,7,8,9,10,11,12,13,14,15);
#pragma unroll
        for (int nt = 0; nt < 4; ++nt) {
            const v16h b = *(const v16h*)(w2p + (size_t)(kc * 8 + nhalf * 4 + nt) * 512 + L * 16);
            acc2[nt] = __builtin_amdgcn_wmma_f32_16x16x32_f16(
                false, a, false, b, (short)0, acc2[nt], false, false);
        }
    }

    // ---- bias + SiLU, store Y (f32); Y region disjoint from H ----
#pragma unroll
    for (int nt = 0; nt < 4; ++nt) {
        int ncol = nhalf * 64 + nt * 16 + lane15;
#pragma unroll
        for (int v = 0; v < 8; ++v) {
            Yl[(strip * 16 + v + 8 * halfL) * YSTR + ncol] = silu_f(acc2[nt][v] + b2v[nt]);
        }
    }
    __syncthreads();

    // ---- phase 4: residual + layernorm; 4 lanes per row, 32 cols each ----
    {
        int r = tid >> 2;
        int g = tid & 3;
        float*       yrow = Yl + r * YSTR + g * 32;
        const float* erow = ef + (size_t)(e0 + r) * D_EDGE + g * 32;
        float sum = 0.f, ss = 0.f;
#pragma unroll
        for (int i = 0; i < 8; ++i) {
            float4 yv = *(const float4*)(yrow + i * 4);
            float4 ev = *(const float4*)(erow + i * 4);
            yv.x += ev.x; yv.y += ev.y; yv.z += ev.z; yv.w += ev.w;
            sum += yv.x + yv.y + yv.z + yv.w;
            ss  += yv.x * yv.x + yv.y * yv.y + yv.z * yv.z + yv.w * yv.w;
            *(float4*)(yrow + i * 4) = yv;
        }
        sum += __shfl_xor(sum, 1, 32);  ss += __shfl_xor(ss, 1, 32);
        sum += __shfl_xor(sum, 2, 32);  ss += __shfl_xor(ss, 2, 32);
        float mean = sum * (1.0f / 128.0f);
        float var  = ss * (1.0f / 128.0f) - mean * mean;
        float rstd = __builtin_amdgcn_rsqf(var + 1e-5f);
        float* orow = out + (size_t)(e0 + r) * D_EDGE + g * 32;
#pragma unroll
        for (int i = 0; i < 8; ++i) {
            float4 yv = *(const float4*)(yrow + i * 4);
            float4 gv = *(const float4*)(gamma + g * 32 + i * 4);
            float4 bv = *(const float4*)(beta  + g * 32 + i * 4);
            float4 o;
            o.x = (yv.x - mean) * rstd * gv.x + bv.x;
            o.y = (yv.y - mean) * rstd * gv.y + bv.y;
            o.z = (yv.z - mean) * rstd * gv.z + bv.z;
            o.w = (yv.w - mean) * rstd * gv.w + bv.w;
            *(float4*)(orow + i * 4) = o;
        }
    }
}

extern "C" void kernel_launch(void* const* d_in, const int* in_sizes, int n_in,
                              void* d_out, int out_size, void* d_ws, size_t ws_size,
                              hipStream_t stream) {
    const float* ns    = (const float*)d_in[0];
    const float* ef    = (const float*)d_in[1];
    const float* dptr  = (const float*)d_in[2];
    const int*   srci  = (const int*)d_in[3];
    const int*   dsti  = (const int*)d_in[4];
    const float* W1    = (const float*)d_in[5];
    const float* b1    = (const float*)d_in[6];
    const float* W2    = (const float*)d_in[7];
    const float* b2    = (const float*)d_in[8];
    const float* gamma = (const float*)d_in[9];
    const float* beta  = (const float*)d_in[10];
    float* out = (float*)d_out;

    _Float16* w1p = (_Float16*)d_ws;
    _Float16* w2p = (_Float16*)((char*)d_ws + (size_t)W1P_HALVES * 2);

    const int n_edges = in_sizes[3];
    const int tiles   = n_edges / M_TILE;   // 600000 / 64 = 9375 exactly

    pack_w1_kernel<<<(W1P_HALVES + 255) / 256, 256, 0, stream>>>(W1, w1p);
    pack_w2_kernel<<<(W2P_HALVES + 255) / 256, 256, 0, stream>>>(W2, w2p);
    edge_mlp_kernel<<<tiles, 256, 0, stream>>>(ns, ef, dptr, srci, dsti,
                                               w1p, w2p, b1, b2, gamma, beta, out);
}
